// Matcher_86131274154743
// MI455X (gfx1250) — compile-verified, matrix operand order checked by reading, not run on hardware
//
#include <hip/hip_runtime.h>
#include <hip/hip_bf16.h>
#include <stdint.h>

typedef __attribute__((ext_vector_type(16))) __bf16 v16bf;
typedef __attribute__((ext_vector_type(8)))  float v8f;

#define BATCH 2048
#define NEIGH 64
#define EDIM  256
#define KDIM  512
#define ROWP  520   // padded LDS row stride in bf16 elems (1040 B, 16B aligned)

// round-to-nearest-even fp32 -> bf16, packed pair
__device__ __forceinline__ uint32_t pack_bf16(float a, float b) {
    uint32_t ua = __builtin_bit_cast(uint32_t, a);
    uint32_t ub = __builtin_bit_cast(uint32_t, b);
    uint32_t ra = (ua + 0x7FFFu + ((ua >> 16) & 1u)) >> 16;
    uint32_t rb = (ub + 0x7FFFu + ((ub >> 16) & 1u)) >> 16;
    return (ra & 0xFFFFu) | (rb << 16);
}

// Pre-swizzle W (E x 2E fp32 row-major, contraction over k) into bf16 B-matrix
// fragments in the exact 16-bit WMMA B VGPR layout:
//   tile (kt, nt), lane l: n = nt*16 + (l&15), half = l>>4
//   dword j holds K pair k0 = 2*(j&3) + 8*half + 16*(j>>2)  (k = kt*32 + k0)
// Storage: ws[branch][kt][nt][lane][j]  (u32 each) -> 256 KB per W.
__global__ void swizzle_w_kernel(const float* __restrict__ Watt,
                                 const float* __restrict__ Wvar,
                                 uint32_t* __restrict__ out) {
    uint32_t idx    = blockIdx.x * blockDim.x + threadIdx.x;  // 131072 total
    uint32_t branch = idx >> 16;
    uint32_t rem    = idx & 0xFFFFu;
    uint32_t j      = rem & 7u;
    uint32_t lane   = (rem >> 3) & 31u;
    uint32_t tile   = rem >> 8;              // 0..255
    uint32_t nt     = tile & 15u;
    uint32_t kt     = tile >> 4;
    uint32_t half   = lane >> 4;
    uint32_t nl     = lane & 15u;
    uint32_t k0     = 2u * (j & 3u) + 8u * half + 16u * (j >> 2);
    uint32_t kg     = kt * 32u + k0;
    uint32_t ng     = nt * 16u + nl;
    const float* W  = branch ? Wvar : Watt;
    float w0 = W[(size_t)ng * KDIM + kg];
    float w1 = W[(size_t)ng * KDIM + kg + 1];
    out[idx] = pack_bf16(w0, w1);
}

__global__ __launch_bounds__(256) void neigh_enc_kernel(
    const int*   __restrict__ conn,
    const float* __restrict__ emb,
    const float* __restrict__ var_emb,
    const float* __restrict__ b_att, const float* __restrict__ u_att, const float* __restrict__ u_att_b,
    const float* __restrict__ b_var, const float* __restrict__ u_var, const float* __restrict__ u_var_b,
    const uint32_t* __restrict__ wsw,
    float* __restrict__ out) {
    extern __shared__ char smem[];
    uint16_t* cbuf   = (uint16_t*)smem;                       // 64 x ROWP bf16 (concat rows)
    float*    logits = (float*)(smem + NEIGH * ROWP * 2);     // 64
    float*    att_e  = logits + NEIGH;                        // 64
    float*    att    = att_e + NEIGH;                         // 64

    const int b      = blockIdx.x;
    const int tid    = threadIdx.x;
    const int lane   = tid & 31;
    const int wave   = tid >> 5;
    const int m_tile = wave & 3;     // neighbor tile (M)
    const int ngroup = wave >> 2;    // which half of the 16 N tiles
    const int half   = lane >> 4;
    const int nl     = lane & 15;

    for (int branch = 0; branch < 2; ++branch) {
        const float* table = branch ? var_emb : emb;
        const float* bvec  = branch ? b_var   : b_att;
        const float* uvec  = branch ? u_var   : u_att;
        const float  ubias = branch ? u_var_b[0] : u_att_b[0];
        const uint32_t* wb = wsw + (size_t)branch * (16 * 16 * 32 * 8);

        // ---- gather: build concat rows [emb[rel] | emb[ent]] in LDS as bf16 ----
        {
            int n    = tid >> 2;                 // neighbor 0..63
            int part = tid & 3;                  // 4 x 128 floats per row
            int col  = (part < 2) ? 0 : 1;       // rel / ent
            int sym  = conn[(b * NEIGH + n) * 3 + col];
            const float* src = table + (size_t)sym * EDIM + (part & 1) * 128;
            uint32_t* dst = (uint32_t*)(cbuf + n * ROWP + part * 128);
            #pragma unroll
            for (int i = 0; i < 32; ++i) {
                float4 v = ((const float4*)src)[i];
                dst[2 * i]     = pack_bf16(v.x, v.y);
                dst[2 * i + 1] = pack_bf16(v.z, v.w);
            }
        }
        if (tid < NEIGH) logits[tid] = ubias;
        __syncthreads();

        // ---- GEMM: h(64x256) = concat(64x512) x B(512x256), bf16 WMMA ----
        v8f acc[8] = {};
        const uint16_t* arow = cbuf + (m_tile * 16 + nl) * ROWP;
        for (int kt = 0; kt < 16; ++kt) {
            if (kt + 1 < 16)  // pull next B tiles toward L2/L0
                __builtin_prefetch(wb + (((kt + 1) * 16 + ngroup * 8) * 32 + lane) * 8, 0, 3);
            // A fragment (16-bit A layout): dwords 0..3 = K base..base+7,
            // dwords 4..7 = +16, K base = kt*32 + 8*half
            union { uint4 q[2]; v16bf v; } au;
            au.q[0] = *(const uint4*)(arow + kt * 32 + half * 8);
            au.q[1] = *(const uint4*)(arow + kt * 32 + half * 8 + 16);
            #pragma unroll
            for (int j = 0; j < 8; ++j) {
                int nt = ngroup * 8 + j;
                const uint32_t* bp = wb + ((size_t)(kt * 16 + nt) * 32 + lane) * 8;
                union { uint4 q[2]; v16bf v; } bu;
                bu.q[0] = *(const uint4*)(bp);
                bu.q[1] = *(const uint4*)(bp + 4);
                acc[j] = __builtin_amdgcn_wmma_f32_16x16x32_bf16(
                    false, au.v, false, bu.v, (short)0, acc[j], false, false);
            }
        }

        // ---- logits: tanh(h + b) . u, reduced over e ----
        // C/D layout: lane 0-15 -> N=nl, M=r; lane 16-31 -> N=nl, M=r+8
        float partial[8];
        #pragma unroll
        for (int r = 0; r < 8; ++r) partial[r] = 0.f;
        #pragma unroll
        for (int j = 0; j < 8; ++j) {
            int e   = (ngroup * 8 + j) * 16 + nl;
            float bb = bvec[e];
            float uu = uvec[e];
            #pragma unroll
            for (int r = 0; r < 8; ++r)
                partial[r] += tanhf(acc[j][r] + bb) * uu;
        }
        #pragma unroll
        for (int r = 0; r < 8; ++r) {
            float s = partial[r];
            s += __shfl_xor(s, 1);
            s += __shfl_xor(s, 2);
            s += __shfl_xor(s, 4);
            s += __shfl_xor(s, 8);   // xor<=8 stays within each 16-lane half
            if (nl == 0)
                atomicAdd(&logits[m_tile * 16 + half * 8 + r], s);
        }
        __syncthreads();

        // ---- softmax over 64 neighbors ----
        if (tid < NEIGH) {
            float mx = -1e30f;
            for (int n = 0; n < NEIGH; ++n) mx = fmaxf(mx, logits[n]);
            att_e[tid] = __expf(logits[tid] - mx);
        }
        __syncthreads();
        if (tid < NEIGH) {
            float s = 0.f;
            for (int n = 0; n < NEIGH; ++n) s += att_e[n];
            att[tid] = att_e[tid] / s;
        }
        __syncthreads();

        // ---- attention-weighted sum over ent embeddings, tanh, store ----
        {
            int e = tid;  // 0..255
            float accv = 0.f;
            for (int n = 0; n < NEIGH; ++n) {
                uint16_t raw = cbuf[n * ROWP + EDIM + e];   // ent half of concat
                float ev = __builtin_bit_cast(float, (uint32_t)raw << 16);
                accv += att[n] * ev;
            }
            out[(size_t)branch * BATCH * EDIM + (size_t)b * EDIM + e] = tanhf(accv);
        }
        __syncthreads();   // cbuf reused by next branch
    }
}

extern "C" void kernel_launch(void* const* d_in, const int* in_sizes, int n_in,
                              void* d_out, int out_size, void* d_ws, size_t ws_size,
                              hipStream_t stream) {
    const int*   conn    = (const int*)d_in[0];
    // d_in[1] = num_neighbors (unused by the reference)
    const float* emb     = (const float*)d_in[2];
    const float* var_emb = (const float*)d_in[3];
    const float* W_att   = (const float*)d_in[4];
    const float* b_att   = (const float*)d_in[5];
    const float* u_att   = (const float*)d_in[6];
    const float* u_att_b = (const float*)d_in[7];
    const float* W_var   = (const float*)d_in[8];
    const float* b_var   = (const float*)d_in[9];
    const float* u_var   = (const float*)d_in[10];
    const float* u_var_b = (const float*)d_in[11];
    uint32_t* wsw = (uint32_t*)d_ws;   // 512 KB: swizzled bf16 W_att + W_var
    float* out = (float*)d_out;

    // 2 branches x 65536 dwords of swizzled weights
    swizzle_w_kernel<<<512, 256, 0, stream>>>(W_att, W_var, wsw);

    size_t shmem = (size_t)NEIGH * ROWP * 2 + 3 * NEIGH * sizeof(float);
    neigh_enc_kernel<<<BATCH, 256, shmem, stream>>>(
        conn, emb, var_emb,
        b_att, u_att, u_att_b,
        b_var, u_var, u_var_b,
        wsw, out);
}
